// BinaryConv2d_64965675319701
// MI455X (gfx1250) — matrix operand order repaired and checked by reading, not compiled, not run on hardware
//
#include <hip/hip_runtime.h>

// ---- CDNA5 vector types ----
typedef __attribute__((ext_vector_type(16))) __bf16 bf16x16;
typedef __attribute__((ext_vector_type(8)))  float  f32x8;
typedef __attribute__((ext_vector_type(4)))  unsigned int u32x4;
typedef __attribute__((ext_vector_type(8)))  int          i32x8;
typedef __attribute__((ext_vector_type(4)))  int          i32x4;

#define C_IN   256
#define C_OUT  256
#define HWSZ   3136                 // 56*56
#define WDIM   56
#define KTOT   2304                 // 9*256, reordered as k = khw*256 + ci
#define BK     32                   // K per WMMA step
#define BCO    128                  // block tile: output channels (GEMM M)
#define BPIX   128                  // block tile: pixels         (GEMM N)
#define NTOT   (32 * HWSZ)          // 100352 flattened (n,h,w) pixels
#define ROWB   80                   // LDS row stride: 64B data + 16B pad (conflict-free)

// ---- one-time: sign(W) -> bf16, reorder OIHW [co][ci*9+khw] -> [co][khw*256+ci] ----
__global__ __launch_bounds__(256)
void prep_weights_kernel(const float* __restrict__ w, unsigned short* __restrict__ w2) {
    const int t   = blockIdx.x * 256 + threadIdx.x;    // 0 .. 256*2304-1
    const int co  = t / KTOT;
    const int rem = t - co * KTOT;
    const int khw = rem >> 8;
    const int ci  = rem & 255;
    const float v = w[(size_t)co * KTOT + ci * 9 + khw];
    const float s = (v > 0.f) ? 1.f : ((v < 0.f) ? -1.f : 0.f);
    const __bf16 b = (__bf16)s;
    w2[t] = __builtin_bit_cast(unsigned short, b);
}

__global__ __launch_bounds__(256)
void binconv_wmma_kernel(const float* __restrict__ x,
                         const unsigned short* __restrict__ w2,
                         float* __restrict__ out) {
    __shared__ __align__(16) unsigned char ldsA[BCO  * ROWB]; // sign(W) bf16 [co][k]
    __shared__ __align__(16) unsigned char ldsB[BPIX * ROWB]; // im2col x bf16 [pix][k]

    const int tid  = threadIdx.x;
    const int wave = tid >> 5;            // 8 wave32 waves
    const int lane = tid & 31;
    const int co0  = blockIdx.y * BCO;
    const int gp0  = blockIdx.x * BPIX;   // flattened (n,h,w) pixel base

    // B-tile loader role: one pixel, 16 consecutive ci per thread
    const int bp  = tid & 127;            // pixel within tile
    const int kg  = tid >> 7;             // 0..1 -> ci group of 16
    const int g   = gp0 + bp;
    const int gn  = g / HWSZ;             // image index
    const int gp  = g - gn * HWSZ;
    const int ph  = gp / WDIM;
    const int pw  = gp - ph * WDIM;

    f32x8 acc[8] = {};

    const int halfsel = (lane < 16) ? 0 : 1;
    const unsigned aoff = (unsigned)((wave * 16 + (lane & 15)) * ROWB + halfsel * 32);
    const unsigned boff = (unsigned)((lane & 15) * ROWB + halfsel * 32);

    // Uniform pieces of the TDM descriptor (2D tile of pre-signed bf16 weights)
    const unsigned lds_a = (unsigned)(size_t)(void*)ldsA;  // low 32 bits = LDS offset

    for (int k0 = 0; k0 < KTOT; k0 += BK) {
        // ---- stage A via Tensor Data Mover: w2[co0..+128][k0..+32] -> ldsA (padded rows) ----
        if (wave == 0) {
            unsigned long long ga =
                (unsigned long long)(size_t)(w2 + (size_t)co0 * KTOT + k0);
            u32x4 g0;
            g0[0] = 1u;                                   // count=1, user descriptor
            g0[1] = lds_a;                                // lds_addr
            g0[2] = (unsigned)ga;                         // global_addr[31:0]
            g0[3] = (unsigned)((ga >> 32) & 0x01FFFFFFull) | 0x80000000u; // addr[56:32] | type=2
            i32x8 g1;
            g1[0] = (1 << 16) | (1 << 20) | (3 << 22) | (3 << 25); // data_size=2B, pad_en, every 16DW pad 4DW
            g1[1] = (int)((KTOT & 0xFFFF) << 16);         // tensor_dim0[15:0]
            g1[2] = (int)((KTOT >> 16) | ((C_OUT & 0xFFFF) << 16)); // tensor_dim0 hi | tensor_dim1 lo
            g1[3] = (int)((BK & 0xFFFF) << 16);           // tensor_dim1 hi=0 | tile_dim0=32
            g1[4] = BCO;                                  // tile_dim1=128, tile_dim2=0
            g1[5] = KTOT;                                 // tensor_dim0_stride[31:0]
            g1[6] = 0;
            g1[7] = 0;
            i32x4 z4 = {0, 0, 0, 0};
            i32x8 z8 = {0, 0, 0, 0, 0, 0, 0, 0};
            __builtin_amdgcn_tensor_load_to_lds(g0, g1, z4, z4, z8, 0);
        }

        // ---- stage B: im2col gather (overlapped with the TDM), (kh,kw) constant per K-step ----
        {
            const int khw = k0 >> 8;                      // uniform
            const int kh  = khw / 3;
            const int kw  = khw - kh * 3;
            const int hh  = ph + kh - 1;
            const int ww  = pw + kw - 1;
            const int ci0 = (k0 & 255) + kg * 16;
            __bf16 tmp[16] __attribute__((aligned(16)));
            if ((unsigned)hh < (unsigned)WDIM && (unsigned)ww < (unsigned)WDIM) {
                const float* src =
                    x + ((size_t)(gn * C_IN + ci0) * HWSZ + hh * WDIM + ww);
#pragma unroll
                for (int j = 0; j < 16; ++j) tmp[j] = (__bf16)src[(size_t)j * HWSZ];
            } else {
#pragma unroll
                for (int j = 0; j < 16; ++j) tmp[j] = (__bf16)0.f;
            }
            uint4* dst = (uint4*)(ldsB + bp * ROWB + kg * 32);
            dst[0] = *(const uint4*)&tmp[0];
            dst[1] = *(const uint4*)&tmp[8];
        }

        if (wave == 0) __builtin_amdgcn_s_wait_tensorcnt(0);
        __syncthreads();

        // ---- fragments + 8x v_wmma_f32_16x16x32_bf16 per wave ----
        bf16x16 afrag;
        {
            uint4* ap = (uint4*)&afrag;
            ap[0] = *(const uint4*)(ldsA + aoff);
            ap[1] = *(const uint4*)(ldsA + aoff + 16);
        }
#pragma unroll
        for (int sub = 0; sub < 8; ++sub) {
            bf16x16 bfrag;
            uint4* bpv = (uint4*)&bfrag;
            const unsigned char* bsrc = ldsB + sub * 16 * ROWB + boff;
            bpv[0] = *(const uint4*)(bsrc);
            bpv[1] = *(const uint4*)(bsrc + 16);
            acc[sub] = __builtin_amdgcn_wmma_f32_16x16x32_bf16(
                false, afrag, false, bfrag, (short)0, acc[sub], false, false);
        }
        __syncthreads();
    }

    // ---- store: lanes -> consecutive pixels, contiguous 64B runs per co row ----
    const int co_base = co0 + wave * 16 + halfsel * 8;
#pragma unroll
    for (int sub = 0; sub < 8; ++sub) {
        const int gcol = gp0 + sub * 16 + (lane & 15);
        const int n    = gcol / HWSZ;
        const int p    = gcol - n * HWSZ;
#pragma unroll
        for (int r = 0; r < 8; ++r) {
            out[((size_t)(n * C_OUT + co_base + r)) * HWSZ + p] = acc[sub][r];
        }
    }
}

extern "C" void kernel_launch(void* const* d_in, const int* in_sizes, int n_in,
                              void* d_out, int out_size, void* d_ws, size_t ws_size,
                              hipStream_t stream) {
    (void)in_sizes; (void)n_in; (void)out_size; (void)ws_size;
    const float* x = (const float*)d_in[0];         // (32,256,56,56) f32
    const float* w = (const float*)d_in[1];         // (256,256,3,3) f32
    float* out = (float*)d_out;
    unsigned short* w2 = (unsigned short*)d_ws;     // 256*2304 bf16 = 1.18 MB

    // 1) binarize + reorder weights (k = khw*256 + ci), bf16
    prep_weights_kernel<<<dim3((C_OUT * KTOT) / 256), dim3(256), 0, stream>>>(w, w2);

    // 2) implicit-GEMM binary conv via WMMA + TDM
    dim3 grid(NTOT / BPIX, C_OUT / BCO, 1);          // 784 x 2 blocks
    binconv_wmma_kernel<<<grid, dim3(256), 0, stream>>>(x, w2, out);
}
